// Attention_8564164789067
// MI455X (gfx1250) — matrix operand order repaired and checked by reading, not compiled, not run on hardware
//
#include <hip/hip_runtime.h>
#include <hip/hip_bf16.h>
#include <stdint.h>

typedef __attribute__((ext_vector_type(16))) __bf16 v16bf;
typedef __attribute__((ext_vector_type(8)))  float  v8f;
typedef unsigned short u16;
typedef unsigned int   u32;

struct Fragbits { uint4 lo, hi; };

__device__ __forceinline__ v16bf frag_cast(Fragbits f) {
  union U { Fragbits b; v16bf v; } u; u.b = f; return u.v;
}
__device__ __forceinline__ u16 f2bf(float x) {
  u32 u = __float_as_uint(x);
  u32 r = u + 0x7FFFu + ((u >> 16) & 1u);   // round-to-nearest-even
  return (u16)(r >> 16);
}
__device__ __forceinline__ float bf2f(u16 h) {
  return __uint_as_float(((u32)h) << 16);
}

// async DMA: global -> LDS, 16 bytes, tracked by ASYNCcnt
__device__ __forceinline__ void async_ld_b128(u32 lds_off, const u16* g) {
  asm volatile("global_load_async_to_lds_b128 %0, %1, off"
               :: "v"(lds_off), "v"(g) : "memory");
}
__device__ __forceinline__ void wait_async0() {
  asm volatile("s_wait_asynccnt 0x0" ::: "memory");
}

// ---------------- elementwise f32 -> bf16 ----------------
__global__ __launch_bounds__(256) void cvt_bf16_kernel(const float* __restrict__ in,
                                                       u16* __restrict__ out, int n) {
  int i = blockIdx.x * blockDim.x + threadIdx.x;
  if (i < n) out[i] = f2bf(in[i]);
}

// ---------------- transpose (RxC f32) -> (CxR bf16) ----------------
__global__ __launch_bounds__(256) void transpose_cvt_kernel(const float* __restrict__ in,
                                                            u16* __restrict__ out,
                                                            int R, int Cc) {
  int t = blockIdx.x * blockDim.x + threadIdx.x;
  if (t >= R * Cc) return;
  int c = t / R, r = t % R;                 // out[c*R + r] = in[r*Cc + c]
  out[t] = f2bf(in[(size_t)r * Cc + c]);
}

// ---------------- tiled bf16 WMMA GEMM: C = A(MxK) * Bt(KxN) ----------------
// block = 256 thr = 8 waves; block tile 128x128, wave tile 64x32.
// K loop in steps of 32; A/B slices double-buffered in LDS, filled by
// GLOBAL_LOAD_ASYNC_TO_LDS_B128 (ASYNCcnt), overlapped with WMMA.
template <bool OUT_BF16>
__global__ __launch_bounds__(256) void gemm_bf16_kernel(const u16* __restrict__ A,
                                                        const u16* __restrict__ Bt,
                                                        void* __restrict__ Out,
                                                        const float* __restrict__ bias,
                                                        int M, int N, int K) {
  __shared__ __align__(16) u16 sA[2][128 * 40];   // 128 rows, stride 40 halfs (80 B)
  __shared__ __align__(16) u16 sB[2][32 * 136];   // 32 k-rows, stride 136 halfs (272 B)

  const int tid  = threadIdx.x;
  const int lane = tid & 31, wv = tid >> 5;
  const int m0 = blockIdx.y * 128;
  const int n0 = blockIdx.x * 128;
  const int wm = wv & 1;        // 2 waves over M (64 rows each)
  const int wn = wv >> 1;       // 4 waves over N (32 cols each)
  const int g = lane >> 4, rr = lane & 15;

  auto issue = [&](int pb, int k0) {
    u32 aBase = (u32)(uintptr_t)(&sA[pb][0]);
    u32 bBase = (u32)(uintptr_t)(&sB[pb][0]);
#pragma unroll
    for (int i = 0; i < 2; i++) {           // A: 128 rows x 4 chunks = 512 x 16B
      int q = tid + i * 256;
      int row = q >> 2, cs = q & 3;
      int gr = m0 + row; if (gr >= M) gr = M - 1;
      async_ld_b128(aBase + (u32)(row * 40 + cs * 8) * 2,
                    A + (size_t)gr * K + k0 + cs * 8);
    }
#pragma unroll
    for (int i = 0; i < 2; i++) {           // B: 32 k-rows x 16 chunks = 512 x 16B
      int q = tid + i * 256;
      int kk = q >> 4, cs = q & 15;
      async_ld_b128(bBase + (u32)(kk * 136 + cs * 8) * 2,
                    Bt + (size_t)(k0 + kk) * N + n0 + cs * 8);
    }
  };

  v8f zero = {0.f,0.f,0.f,0.f,0.f,0.f,0.f,0.f};
  v8f acc[4][2];
#pragma unroll
  for (int i = 0; i < 4; i++)
#pragma unroll
    for (int j = 0; j < 2; j++) acc[i][j] = zero;

  issue(0, 0);
  const int nk = K / 32;
  for (int t = 0; t < nk; t++) {
    int pb = t & 1;
    wait_async0();                // slice t landed (slice t+1 not yet issued)
    __syncthreads();              // everyone sees it; everyone done reading pb^1
    if (t + 1 < nk) issue(pb ^ 1, (t + 1) * 32);   // DMA overlaps WMMA below

    v16bf bfr[2];
#pragma unroll
    for (int j = 0; j < 2; j++) {
      const u16* p = &sB[pb][wn * 32 + j * 16 + rr];
      Fragbits f; u16* fp = (u16*)&f;
#pragma unroll
      for (int e = 0; e < 16; e++) fp[e] = p[(e < 8 ? g * 8 + e : 16 + g * 8 + (e - 8)) * 136];
      bfr[j] = frag_cast(f);
    }
#pragma unroll
    for (int i = 0; i < 4; i++) {
      const u16* p = &sA[pb][(wm * 64 + i * 16 + rr) * 40 + g * 8];
      Fragbits f; f.lo = *(const uint4*)p; f.hi = *(const uint4*)(p + 16);
      v16bf afr = frag_cast(f);
#pragma unroll
      for (int j = 0; j < 2; j++)
        acc[i][j] = __builtin_amdgcn_wmma_f32_16x16x32_bf16(
            false, afr, false, bfr[j], (short)0, acc[i][j], false, false);
    }
  }

#pragma unroll
  for (int i = 0; i < 4; i++)
#pragma unroll
    for (int j = 0; j < 2; j++) {
      int n = n0 + wn * 32 + j * 16 + rr;
#pragma unroll
      for (int r = 0; r < 8; r++) {
        int m = m0 + wm * 64 + i * 16 + r + 8 * g;
        if (m < M) {
          float v = acc[i][j][r];
          if (OUT_BF16) ((u16*)Out)[(size_t)m * N + n] = f2bf(v);
          else          ((float*)Out)[(size_t)m * N + n] = v + bias[n];
        }
      }
    }
}

// ---------------- windowed attention + LePE ----------------
// grid = 1024 windows * 12 heads; block = 256 thr (8 waves).
__global__ __launch_bounds__(256) void attn_kernel(const u16* __restrict__ qkv,
                                                   const float* __restrict__ conv_w,
                                                   const float* __restrict__ conv_b,
                                                   u16* __restrict__ outbf) {
  __shared__ __align__(16) u16  kT[64 * 72];   // channel-major K
  __shared__ __align__(16) u16  vS[64 * 72];   // token-major V (rows>=49 zero)
  __shared__ __align__(16) u16  aB[64 * 72];   // bf16 softmaxed attn (cols>=49 zero)
  __shared__ __align__(16) float aF[64 * 65];  // fp32 logits
  __shared__ float cw[64 * 9];
  __shared__ float cb[64];

  const int tid  = threadIdx.x;
  const int lane = tid & 31, wv = tid >> 5;
  const int w = blockIdx.x / 12, h = blockIdx.x % 12;
  const int b = w >> 6, hb = (w >> 3) & 7, wb = w & 7;

  auto grow = [&](int p) -> size_t {          // window pos -> qkv row offset (elems)
    int n = 1 + (hb * 7 + p / 7) * 56 + (wb * 7 + p % 7);
    return ((size_t)b * 3137 + n) * 2304;
  };

  for (int e = tid; e < 64 * 64; e += 256) {
    int p = e >> 6, c = e & 63;
    if (p < 49) {
      size_t base = grow(p) + h * 64 + c;
      kT[c * 72 + p] = qkv[base + 768];
      vS[p * 72 + c] = qkv[base + 1536];
    } else {
      kT[c * 72 + p] = 0;
      vS[p * 72 + c] = 0;
    }
  }
  for (int e = tid; e < 64 * 72; e += 256) aB[e] = 0;
  for (int e = tid; e < 64 * 9;  e += 256) cw[e] = conv_w[(size_t)(h * 64) * 9 + e];
  if (tid < 64) cb[tid] = conv_b[h * 64 + tid];
  __syncthreads();

  const int mi  = wv & 3;          // M tile (query rows)
  const int njb = (wv >> 2) * 2;   // two N tiles per wave
  const int g = lane >> 4, rr = lane & 15;
  v8f zero = {0.f,0.f,0.f,0.f,0.f,0.f,0.f,0.f};

  // ---- Q * K^T ----
  v8f accA[2] = {zero, zero};
#pragma unroll
  for (int s = 0; s < 2; s++) {                       // K dim 64 = 2 x 32
    int t = mi * 16 + rr; if (t > 48) t = 48;         // clamp pad rows
    const u16* p = qkv + grow(t) + h * 64 + s * 32 + g * 8;
    Fragbits fa; fa.lo = *(const uint4*)p; fa.hi = *(const uint4*)(p + 16);
    v16bf a = frag_cast(fa);
#pragma unroll
    for (int j = 0; j < 2; j++) {
      int cch = s * 32;                               // lane -> token column of K^T, elems over channels
      const u16* q = &kT[cch * 72 + (njb + j) * 16 + rr];
      Fragbits fb; u16* fq = (u16*)&fb;
#pragma unroll
      for (int e = 0; e < 16; e++) fq[e] = q[(e < 8 ? g * 8 + e : 16 + g * 8 + (e - 8)) * 72];
      accA[j] = __builtin_amdgcn_wmma_f32_16x16x32_bf16(
          false, a, false, frag_cast(fb), (short)0, accA[j], false, false);
    }
  }
#pragma unroll
  for (int j = 0; j < 2; j++) {
    int n = (njb + j) * 16 + rr;
#pragma unroll
    for (int r = 0; r < 8; r++) aF[(mi * 16 + r + 8 * g) * 65 + n] = accA[j][r];
  }
  __syncthreads();

  // ---- softmax (rows 0..48, cols 0..48) ----
  if (tid < 49) {
    const float scale = 0.125f;                       // 64^-0.5
    float mx = -3.0e38f;
    for (int c = 0; c < 49; c++) {
      float v = aF[tid * 65 + c] * scale; aF[tid * 65 + c] = v; if (v > mx) mx = v;
    }
    float sum = 0.f;
    for (int c = 0; c < 49; c++) {
      float e = __expf(aF[tid * 65 + c] - mx); aF[tid * 65 + c] = e; sum += e;
    }
    float inv = 1.0f / sum;
    for (int c = 0; c < 49; c++) aB[tid * 72 + c] = f2bf(aF[tid * 65 + c] * inv);
  }
  __syncthreads();

  // ---- attn * V ----
  v8f accO[2] = {zero, zero};
#pragma unroll
  for (int s = 0; s < 2; s++) {                       // K dim 64 (cols 49..63 are zero)
    int l = mi * 16 + rr;
    const u16* p = &aB[l * 72 + s * 32 + g * 8];
    Fragbits fa; fa.lo = *(const uint4*)p; fa.hi = *(const uint4*)(p + 16);
    v16bf a = frag_cast(fa);
#pragma unroll
    for (int j = 0; j < 2; j++) {
      int kt = s * 32;                                // lane -> channel column of V, elems over tokens
      const u16* q = &vS[kt * 72 + (njb + j) * 16 + rr];
      Fragbits fb; u16* fq = (u16*)&fb;
#pragma unroll
      for (int e = 0; e < 16; e++) fq[e] = q[(e < 8 ? g * 8 + e : 16 + g * 8 + (e - 8)) * 72];
      accO[j] = __builtin_amdgcn_wmma_f32_16x16x32_bf16(
          false, a, false, frag_cast(fb), (short)0, accO[j], false, false);
    }
  }

  // ---- add LePE (3x3 depthwise in-window) and store; window-blocked token order ----
#pragma unroll
  for (int j = 0; j < 2; j++) {
    int n = (njb + j) * 16 + rr;                      // channel within head
#pragma unroll
    for (int r = 0; r < 8; r++) {
      int m = mi * 16 + r + 8 * g;                    // token within window
      if (m < 49) {
        int ii = m / 7, jj = m % 7;
        float lepe = cb[n];
#pragma unroll
        for (int di = -1; di <= 1; di++)
#pragma unroll
          for (int dj = -1; dj <= 1; dj++) {
            int yi = ii + di, xj = jj + dj;
            if (yi >= 0 && yi < 7 && xj >= 0 && xj < 7)
              lepe += bf2f(vS[(yi * 7 + xj) * 72 + n]) * cw[n * 9 + (di + 1) * 3 + (dj + 1)];
          }
        float val = accO[j][r] + lepe;
        int ntok = 1 + (hb * 8 + wb) * 49 + m;        // reference keeps window order
        outbf[((size_t)b * 3137 + ntok) * 768 + h * 64 + n] = f2bf(val);
      }
    }
  }
}

// ---------------- cls token passthrough (row n=0 = q_cls) ----------------
__global__ __launch_bounds__(256) void cls_copy_kernel(const u16* __restrict__ qkv,
                                                       u16* __restrict__ outbf) {
  int i = blockIdx.x * blockDim.x + threadIdx.x;
  if (i < 16 * 768) {
    int b = i / 768, c = i % 768;
    outbf[(size_t)b * 3137 * 768 + c] = qkv[(size_t)b * 3137 * 2304 + c];
  }
}

extern "C" void kernel_launch(void* const* d_in, const int* in_sizes, int n_in,
                              void* d_out, int out_size, void* d_ws, size_t ws_size,
                              hipStream_t stream) {
  const float* x      = (const float*)d_in[0];
  const float* qkv_w  = (const float*)d_in[1];
  const float* conv_w = (const float*)d_in[2];
  const float* conv_b = (const float*)d_in[3];
  const float* proj_w = (const float*)d_in[4];
  const float* proj_b = (const float*)d_in[5];

  const int M = 16 * 3137;   // 50192 rows
  const int C = 768, N3 = 2304;

  char* ws = (char*)d_ws;
  u16* x_bf   = (u16*)(ws);                 // 77,094,912 B
  u16* qkvT   = (u16*)(ws + 77094912);      //  3,538,944 B (768 x 2304)
  u16* projT  = (u16*)(ws + 80633856);      //  1,179,648 B (768 x 768)
  u16* qkv_bf = (u16*)(ws + 81813504);      // 231,284,736 B (M x 2304)
  u16* ao_bf  = (u16*)(ws + 313098240);     // 77,094,912 B (M x 768)

  int nx = M * C;
  cvt_bf16_kernel<<<(nx + 255) / 256, 256, 0, stream>>>(x, x_bf, nx);
  transpose_cvt_kernel<<<(N3 * C + 255) / 256, 256, 0, stream>>>(qkv_w, qkvT, N3, C);
  transpose_cvt_kernel<<<(C * C + 255) / 256, 256, 0, stream>>>(proj_w, projT, C, C);

  gemm_bf16_kernel<true><<<dim3(N3 / 128, (M + 127) / 128), 256, 0, stream>>>(
      x_bf, qkvT, qkv_bf, nullptr, M, N3, C);

  attn_kernel<<<16 * 64 * 12, 256, 0, stream>>>(qkv_bf, conv_w, conv_b, ao_bf);
  cls_copy_kernel<<<(16 * 768 + 255) / 256, 256, 0, stream>>>(qkv_bf, ao_bf);

  gemm_bf16_kernel<false><<<dim3(C / 128, (M + 127) / 128), 256, 0, stream>>>(
      ao_bf, projT, d_out, proj_b, M, C, C);
}
